// DNC_62964220559489
// MI455X (gfx1250) — matrix operand order repaired and checked by reading, not compiled, not run on hardware
//
#include <hip/hip_runtime.h>
#include <hip/hip_bf16.h>
#include <math.h>

typedef __attribute__((ext_vector_type(16))) _Float16 v16h;
typedef __attribute__((ext_vector_type(8)))  float    v8f;

#define TT   512
#define BB   64
#define INP  256
#define OUTD 256
#define HH   512
#define NN   2048
#define CC   64
#define G4H  (4*HH)       // 2048
#define EPSF 1e-8f

#define NT_G (G4H/16)     // 128 column tiles of gates
#define KT_X (INP/32)     // 8
#define KT_R (CC/32)      // 2
#define KT_H (HH/32)      // 16
#define MT   (BB/16)      // 4 row tiles

// ---- WMMA fragment index maps (per CDNA5 ISA 7.12.2, 16-bit A 16x32 / B 32x16) ----
__device__ __forceinline__ int kofA(int e, int lane) {
  return ((e < 8) ? e : e + 8) + ((lane & 16) ? 8 : 0);
}
__device__ __forceinline__ int kofB(int e, int lane) {
  return ((e < 8) ? e : e + 8) + ((lane & 16) ? 8 : 0);
}
// inverse of A map: (kk = k%32, m16 = m%16) -> (lane, e)
__device__ __forceinline__ void invA(int kk, int m16, int* lane, int* e) {
  if (kk < 8)       { *lane = m16;      *e = kk;      }
  else if (kk < 16) { *lane = m16 + 16; *e = kk - 8;  }
  else if (kk < 24) { *lane = m16;      *e = kk - 8;  }
  else              { *lane = m16 + 16; *e = kk - 16; }
}

__device__ __forceinline__ v8f wmma_f16(const _Float16* Ap, const _Float16* Bp, v8f acc, int lane) {
  v16h a = *(const v16h*)(Ap + lane * 16);
  v16h b = *(const v16h*)(Bp + lane * 16);
  return __builtin_amdgcn_wmma_f32_16x16x32_f16(false, a, false, b, (short)0, acc, false, false);
}

// ---------------- packing kernels (A-layout for xs; B-layout for weights) ----------------
__global__ void pack_xs(const float* __restrict__ xs, _Float16* __restrict__ dst) {
  long total = (long)TT * MT * KT_X * 512;
  for (long i = blockIdx.x * (long)blockDim.x + threadIdx.x; i < total;
       i += (long)gridDim.x * blockDim.x) {
    int e = i & 15; int lane = (i >> 4) & 31; long tl = i >> 9;
    int kt = tl % KT_X; int mt = (tl / KT_X) % MT; int t = (int)(tl / (KT_X * MT));
    int m = mt * 16 + (lane & 15);
    int k = kt * 32 + kofA(e, lane);
    dst[i] = (_Float16)xs[((long)t * BB + m) * INP + k];
  }
}

// mode 0: plain src[k*NCL + n]
// mode 1: read-head fold: sum_{r<4} W_ih[(INP + 4k + r)*G4H + n]
// mode 2: beta pad to 16 cols: (n==0) ? src[k] : 0
__global__ void pack_w(const float* __restrict__ src, _Float16* __restrict__ dst,
                       int K, int NCL, int mode) {
  int KT = K / 32, NT = NCL / 16;
  long total = (long)KT * NT * 512;
  for (long i = blockIdx.x * (long)blockDim.x + threadIdx.x; i < total;
       i += (long)gridDim.x * blockDim.x) {
    int e = i & 15; int lane = (i >> 4) & 31; long tl = i >> 9;
    int nt = tl % NT; int kt = (int)(tl / NT);
    int n = nt * 16 + (lane & 15);
    int k = kt * 32 + kofB(e, lane);
    float v;
    if (mode == 0)      v = src[(long)k * NCL + n];
    else if (mode == 1) { v = 0.f; for (int r = 0; r < 4; ++r) v += src[(long)(INP + k * 4 + r) * G4H + n]; }
    else                v = (n == 0) ? src[k] : 0.f;
    dst[i] = (_Float16)v;
  }
}

// ---------- fused controller: gates WMMA GEMM + LSTM pointwise + pack h ----------
// Block (mt, jp): 16 rows x 32 LSTM cols; 8 waves = 4 gates x 2 j-subtiles.
// Reads hPin (prev step, packed), writes hPout (packed) -- ping-pong buffers.
__global__ void __launch_bounds__(256) gates_lstm_fused(
    const _Float16* __restrict__ xsP, const _Float16* __restrict__ rP,
    const _Float16* __restrict__ hPin, const _Float16* __restrict__ WxP,
    const _Float16* __restrict__ WrP,  const _Float16* __restrict__ WhP,
    const float* __restrict__ b_lstm,  float* __restrict__ cst,
    _Float16* __restrict__ hPout, int t) {
  __shared__ float ldsG[4][16][32];
  int lane = threadIdx.x & 31;
  int wave = threadIdx.x >> 5;
  int mt = blockIdx.x >> 4;          // 0..3
  int jp = blockIdx.x & 15;          // 0..15  (32 LSTM columns each)
  int gate = wave & 3;               // i,f,g,o
  int jthalf = wave >> 2;            // 0..1
  int jt = jp * 2 + jthalf;          // 0..31
  int nt = gate * 32 + jt;           // tile col in 2048-wide gates

  v8f acc = {};
  const _Float16* xA = xsP + ((long)t * MT + mt) * KT_X * 512;
  #pragma unroll
  for (int kt = 0; kt < KT_X; ++kt)
    acc = wmma_f16(xA + kt * 512, WxP + ((long)kt * NT_G + nt) * 512, acc, lane);
  #pragma unroll
  for (int kt = 0; kt < KT_R; ++kt)
    acc = wmma_f16(rP + (mt * KT_R + kt) * 512, WrP + ((long)kt * NT_G + nt) * 512, acc, lane);
  #pragma unroll
  for (int kt = 0; kt < KT_H; ++kt)
    acc = wmma_f16(hPin + (mt * KT_H + kt) * 512, WhP + ((long)kt * NT_G + nt) * 512, acc, lane);

  int col = nt * 16 + (lane & 15);
  float bias = b_lstm[col];
  int rbase = (lane & 16) ? 8 : 0;
  int lcol = jthalf * 16 + (lane & 15);
  #pragma unroll
  for (int g = 0; g < 8; ++g)
    ldsG[gate][rbase + g][lcol] = acc[g] + bias;
  __syncthreads();

  // LSTM pointwise on the block's 16x32 slice
  #pragma unroll
  for (int q = 0; q < 2; ++q) {
    int e2 = threadIdx.x + q * 256;      // 0..511
    int row = e2 >> 5, lc = e2 & 31;
    int j = jp * 32 + lc;                // 0..511
    int m = mt * 16 + row;               // batch row
    float gi = ldsG[0][row][lc], gf = ldsG[1][row][lc];
    float gg = ldsG[2][row][lc], go = ldsG[3][row][lc];
    float si = 1.f / (1.f + expf(-gi));
    float sf = 1.f / (1.f + expf(-gf));
    float so = 1.f / (1.f + expf(-go));
    int idx = m * HH + j;
    float c = sf * cst[idx] + si * tanhf(gg);
    float h = so * tanhf(c);
    cst[idx] = c;
    int kt = j >> 5, kk = j & 31, plane, pe;
    invA(kk, m & 15, &plane, &pe);
    hPout[((m >> 4) * KT_H + kt) * 512 + plane * 16 + pe] = (_Float16)h;
  }
}

// ---------------- head GEMMs: y_t, key, gen, softplus(beta) ----------------
__global__ void __launch_bounds__(256) heads_gemm(
    const _Float16* __restrict__ hP,  const _Float16* __restrict__ WoP,
    const _Float16* __restrict__ WkP, const _Float16* __restrict__ WgP,
    const _Float16* __restrict__ WbP,
    const float* __restrict__ b_out, const float* __restrict__ b_key,
    const float* __restrict__ b_gen, const float* __restrict__ b_beta,
    float* __restrict__ yout, float* __restrict__ key, float* __restrict__ gen,
    float* __restrict__ beta, int t) {
  int lane = threadIdx.x & 31;
  int wave = threadIdx.x >> 5;
  int tile = blockIdx.x * 8 + wave;
  if (tile >= 100) return;                    // wave-uniform exit
  int mt = tile / 25, sub = tile % 25;
  const _Float16* Wp; int nt, NT, kind;
  if (sub < 16)      { kind = 0; nt = sub;      NT = 16; Wp = WoP; }
  else if (sub < 20) { kind = 1; nt = sub - 16; NT = 4;  Wp = WkP; }
  else if (sub < 24) { kind = 2; nt = sub - 20; NT = 4;  Wp = WgP; }
  else               { kind = 3; nt = 0;        NT = 1;  Wp = WbP; }
  v8f acc = {};
  #pragma unroll
  for (int kt = 0; kt < KT_H; ++kt)
    acc = wmma_f16(hP + (mt * KT_H + kt) * 512, Wp + ((long)kt * NT + nt) * 512, acc, lane);
  int col = nt * 16 + (lane & 15);
  int rbase = mt * 16 + ((lane & 16) ? 8 : 0);
  if (kind == 0) {
    float bias = b_out[col];
    for (int g = 0; g < 8; ++g)
      yout[((long)t * BB + rbase + g) * OUTD + col] = acc[g] + bias;
  } else if (kind == 1) {
    float bias = b_key[col];
    for (int g = 0; g < 8; ++g) key[(rbase + g) * CC + col] = acc[g] + bias;
  } else if (kind == 2) {
    float bias = b_gen[col];
    for (int g = 0; g < 8; ++g) gen[(rbase + g) * CC + col] = acc[g] + bias;
  } else if ((lane & 15) == 0) {
    float bias = b_beta[0];
    for (int g = 0; g < 8; ++g) {
      float x = acc[g] + bias;
      beta[rbase + g] = (x > 20.f) ? x : log1pf(expf(x));
    }
  }
}

// ---------------- memory phase (1024 threads/block, one block per batch elem) ----------
// cosine sim -> softmax over N -> mem += w*key -> r = sum_n w*mem_new; also per-b loss partial
__global__ void __launch_bounds__(1024) memory_kernel(
    float* __restrict__ mem, const float* __restrict__ key, const float* __restrict__ gen,
    const float* __restrict__ beta, float* __restrict__ rvec, _Float16* __restrict__ rP,
    float* __restrict__ lossPart) {
  int b = blockIdx.x;
  int tid = threadIdx.x;
  __shared__ float kbuf[CC];
  __shared__ float simw[NN];
  __shared__ float red[1024];
  __shared__ float4 rr[1024];
  __shared__ float sc[2];
  if (tid < CC) {
    float kv = key[b * CC + tid];
    kbuf[tid] = kv;
    float d = kv - gen[b * CC + tid];
    red[tid] = d * d;
  }
  __syncthreads();
  if (tid == 0) {
    float s = 0.f, l = 0.f;
    for (int i = 0; i < CC; ++i) { s += kbuf[i] * kbuf[i]; l += red[i]; }
    sc[0] = fmaxf(sqrtf(s), EPSF);
    sc[1] = beta[b];
    lossPart[b] = l;
  }
  __syncthreads();
  float kn = sc[0], bet = sc[1];
  int c4 = (tid & 15) * 4;
  int sgrp = tid >> 4;                       // 64 slots in flight
  float k0 = kbuf[c4], k1 = kbuf[c4 + 1], k2 = kbuf[c4 + 2], k3 = kbuf[c4 + 3];
  float* memb = mem + (long)b * NN * CC;
  // pass 1: dot & norm per slot (16-lane butterfly reduce, contiguous float4 loads)
  for (int it = 0; it < NN / 64; ++it) {
    int n = it * 64 + sgrp;
    float4 m4 = *(const float4*)(memb + (long)n * CC + c4);
    float dot = m4.x * k0 + m4.y * k1 + m4.z * k2 + m4.w * k3;
    float nr  = m4.x * m4.x + m4.y * m4.y + m4.z * m4.z + m4.w * m4.w;
    for (int o = 1; o < 16; o <<= 1) { dot += __shfl_xor(dot, o); nr += __shfl_xor(nr, o); }
    if ((tid & 15) == 0) simw[n] = dot / (fmaxf(sqrtf(nr), EPSF) * kn);
  }
  __syncthreads();
  // softmax over N in LDS
  float mx = -1e30f;
  for (int i = tid; i < NN; i += 1024) mx = fmaxf(mx, bet * simw[i]);
  red[tid] = mx; __syncthreads();
  for (int st = 512; st > 0; st >>= 1) { if (tid < st) red[tid] = fmaxf(red[tid], red[tid + st]); __syncthreads(); }
  mx = red[0]; __syncthreads();
  float sum = 0.f;
  for (int i = tid; i < NN; i += 1024) { float ev = expf(bet * simw[i] - mx); simw[i] = ev; sum += ev; }
  red[tid] = sum; __syncthreads();
  for (int st = 512; st > 0; st >>= 1) { if (tid < st) red[tid] += red[tid + st]; __syncthreads(); }
  float inv = 1.f / red[0]; __syncthreads();
  for (int i = tid; i < NN; i += 1024) simw[i] *= inv;
  __syncthreads();
  // pass 2: mem += w * key, and r = sum_n w * mem_new
  float4 racc = make_float4(0.f, 0.f, 0.f, 0.f);
  for (int it = 0; it < NN / 64; ++it) {
    int n = it * 64 + sgrp;
    float w = simw[n];
    float* mp = memb + (long)n * CC + c4;
    float4 m4 = *(const float4*)mp;
    m4.x += w * k0; m4.y += w * k1; m4.z += w * k2; m4.w += w * k3;
    *(float4*)mp = m4;
    racc.x += w * m4.x; racc.y += w * m4.y; racc.z += w * m4.z; racc.w += w * m4.w;
  }
  rr[tid] = racc; __syncthreads();
  if (tid < 16) {
    float4 s = rr[tid];
    for (int g = 1; g < 64; ++g) {
      float4 o = rr[g * 16 + tid];
      s.x += o.x; s.y += o.y; s.z += o.z; s.w += o.w;
    }
    *(float4*)(rvec + b * CC + c4) = s;
    int mt = b >> 4, m16 = b & 15;
    float vals[4] = {s.x, s.y, s.z, s.w};
    for (int q = 0; q < 4; ++q) {
      int k = c4 + q, kt = k >> 5, kk = k & 31, plane, pe;
      invA(kk, m16, &plane, &pe);
      rP[(mt * KT_R + kt) * 512 + plane * 16 + pe] = (_Float16)vals[q];
    }
  }
}

// ---------------- deterministic final loss reduction over all T*B partials ----------------
__global__ void loss_final(const float* __restrict__ lossPart, float* __restrict__ out) {
  __shared__ float red[256];
  float s = 0.f;
  for (int i = threadIdx.x; i < TT * BB; i += 256) s += lossPart[i];
  red[threadIdx.x] = s; __syncthreads();
  for (int st = 128; st > 0; st >>= 1) {
    if (threadIdx.x < st) red[threadIdx.x] += red[threadIdx.x + st];
    __syncthreads();
  }
  if (threadIdx.x == 0) out[0] = red[0] / (float)(BB * CC);
}

extern "C" void kernel_launch(void* const* d_in, const int* in_sizes, int n_in,
                              void* d_out, int out_size, void* d_ws, size_t ws_size,
                              hipStream_t stream) {
  (void)in_sizes; (void)n_in; (void)out_size; (void)ws_size;
  const float* xs     = (const float*)d_in[0];
  const float* W_ih   = (const float*)d_in[1];
  const float* W_hh   = (const float*)d_in[2];
  const float* b_lstm = (const float*)d_in[3];
  const float* W_out  = (const float*)d_in[4];
  const float* b_out  = (const float*)d_in[5];
  const float* W_key  = (const float*)d_in[6];
  const float* b_key  = (const float*)d_in[7];
  const float* W_beta = (const float*)d_in[8];
  const float* b_beta = (const float*)d_in[9];
  const float* W_gen  = (const float*)d_in[10];
  const float* b_gen  = (const float*)d_in[11];
  float* dout = (float*)d_out;

  char* ws = (char*)d_ws;
  size_t off = 0;
  auto alloc = [&](size_t bytes) -> void* {
    off = (off + 255) & ~(size_t)255;
    void* p = ws + off; off += bytes; return p;
  };
  float*     mem   = (float*)alloc((size_t)BB * NN * CC * 4);
  float*     cst   = (float*)alloc((size_t)BB * HH * 4);
  float*     keyb  = (float*)alloc((size_t)BB * CC * 4);
  float*     genb  = (float*)alloc((size_t)BB * CC * 4);
  float*     betab = (float*)alloc((size_t)BB * 4);
  float*     rvec  = (float*)alloc((size_t)BB * CC * 4);
  float*     lossP = (float*)alloc((size_t)TT * BB * 4);
  _Float16*  hPA   = (_Float16*)alloc((size_t)MT * KT_H * 512 * 2);
  _Float16*  hPB   = (_Float16*)alloc((size_t)MT * KT_H * 512 * 2);
  _Float16*  rP    = (_Float16*)alloc((size_t)MT * KT_R * 512 * 2);
  _Float16*  xsP   = (_Float16*)alloc((size_t)TT * MT * KT_X * 512 * 2);
  _Float16*  WxP   = (_Float16*)alloc((size_t)KT_X * NT_G * 512 * 2);
  _Float16*  WrP   = (_Float16*)alloc((size_t)KT_R * NT_G * 512 * 2);
  _Float16*  WhP   = (_Float16*)alloc((size_t)KT_H * NT_G * 512 * 2);
  _Float16*  WoP   = (_Float16*)alloc((size_t)KT_H * 16 * 512 * 2);
  _Float16*  WkP   = (_Float16*)alloc((size_t)KT_H * 4 * 512 * 2);
  _Float16*  WgP   = (_Float16*)alloc((size_t)KT_H * 4 * 512 * 2);
  _Float16*  WbP   = (_Float16*)alloc((size_t)KT_H * 1 * 512 * 2);

  // zero recurrent state (graph-capturable memset nodes)
  hipMemsetAsync(mem, 0, (size_t)BB * NN * CC * 4, stream);
  hipMemsetAsync(cst, 0, (size_t)BB * HH * 4, stream);
  hipMemsetAsync(hPA, 0, (size_t)MT * KT_H * 512 * 2, stream);
  hipMemsetAsync(hPB, 0, (size_t)MT * KT_H * 512 * 2, stream);
  hipMemsetAsync(rP, 0, (size_t)MT * KT_R * 512 * 2, stream);

  // one-time packing (f32 -> f16 WMMA fragment layouts)
  pack_xs<<<4096, 256, 0, stream>>>(xs, xsP);
  pack_w<<<2048, 256, 0, stream>>>(W_ih, WxP, INP, G4H, 0);
  pack_w<<<512, 256, 0, stream>>>(W_ih, WrP, CC, G4H, 1);
  pack_w<<<4096, 256, 0, stream>>>(W_hh, WhP, HH, G4H, 0);
  pack_w<<<512, 256, 0, stream>>>(W_out, WoP, HH, OUTD, 0);
  pack_w<<<128, 256, 0, stream>>>(W_key, WkP, HH, CC, 0);
  pack_w<<<128, 256, 0, stream>>>(W_gen, WgP, HH, CC, 0);
  pack_w<<<32, 256, 0, stream>>>(W_beta, WbP, HH, 16, 2);

  for (int t = 0; t < TT; ++t) {
    const _Float16* hRead = (t & 1) ? hPB : hPA;
    _Float16*       hWrite = (t & 1) ? hPA : hPB;
    gates_lstm_fused<<<64, 256, 0, stream>>>(xsP, rP, hRead, WxP, WrP, WhP,
                                             b_lstm, cst, hWrite, t);
    heads_gemm<<<13, 256, 0, stream>>>(hWrite, WoP, WkP, WgP, WbP,
                                       b_out, b_key, b_gen, b_beta,
                                       dout, keyb, genb, betab, t);
    memory_kernel<<<BB, 1024, 0, stream>>>(mem, keyb, genb, betab, rvec, rP,
                                           lossP + (size_t)t * BB);
  }
  loss_final<<<1, 256, 0, stream>>>(lossP, dout + (size_t)TT * BB * OUTD);
}
